// QscoHead_63221918597666
// MI455X (gfx1250) — matrix operand order repaired and checked by reading, not compiled, not run on hardware
//
#include <hip/hip_runtime.h>
#include <hip/hip_fp16.h>

#define IN_DIM 512
#define LN_EPS 1e-5f

typedef __attribute__((ext_vector_type(16))) _Float16 v16h;
typedef __attribute__((ext_vector_type(8)))  float    v8f;

// XOR lane exchange via ds_swizzle (group-of-32 mode: and=0x1f, or=0, xor=MASK)
template<int MASK>
__device__ __forceinline__ float swz_xor(float x) {
    return __int_as_float(__builtin_amdgcn_ds_swizzle(__float_as_int(x), (MASK << 10) | 0x1f));
}

__device__ __forceinline__ float wave_sum32(float x) {
    x += swz_xor<1>(x);
    x += swz_xor<2>(x);
    x += swz_xor<4>(x);
    x += swz_xor<8>(x);
    x += swz_xor<16>(x);
    return x;
}

// ---- statevector circuit primitives (flat bits 7..3 = lane bits 4..0, bits 2..0 = reg idx) ----
template<int CM, int TM>
__device__ __forceinline__ void cnot_ll(float v[8], int lane) {
    bool cb = (lane & CM) != 0;
    #pragma unroll
    for (int r = 0; r < 8; ++r) {
        float t = swz_xor<TM>(v[r]);
        v[r] = cb ? t : v[r];
    }
}

template<int M>
__device__ __forceinline__ void ry_lane(float v[8], int lane, float c, float s) {
    float ss = (lane & M) ? s : -s;
    #pragma unroll
    for (int r = 0; r < 8; ++r) {
        float p = swz_xor<M>(v[r]);
        v[r] = fmaf(c, v[r], ss * p);
    }
}

template<int BIT>
__device__ __forceinline__ void ry_reg(float v[8], float c, float s) {
    #pragma unroll
    for (int r0 = 0; r0 < 8; ++r0) {
        if (!(r0 & BIT)) {
            const int r1 = r0 | BIT;
            float a0 = v[r0], a1 = v[r1];
            v[r0] = fmaf(c, a0, -s * a1);
            v[r1] = fmaf(s, a0,  c * a1);
        }
    }
}

__device__ __forceinline__ void apply_layers(float v[8], int lane, const float ct[8], const float st[8]) {
    #pragma unroll
    for (int d = 0; d < 2; ++d) {
        // CNOT ring: (q0->q1)(q1->q2)(q2->q3)(q3->q4) lane/lane
        cnot_ll<16, 8>(v, lane);
        cnot_ll<8, 4>(v, lane);
        cnot_ll<4, 2>(v, lane);
        cnot_ll<2, 1>(v, lane);
        {   // (q4->q5): control lane bit0, target reg bit2
            bool cb = (lane & 1) != 0;
            #pragma unroll
            for (int r = 0; r < 4; ++r) {
                float t0 = v[r], t1 = v[r + 4];
                v[r]     = cb ? t1 : t0;
                v[r + 4] = cb ? t0 : t1;
            }
        }
        {   // (q5->q6): swap regs (4,6),(5,7)
            float t = v[4]; v[4] = v[6]; v[6] = t;
            t = v[5]; v[5] = v[7]; v[7] = t;
        }
        {   // (q6->q7): swap regs (2,3),(6,7)
            float t = v[2]; v[2] = v[3]; v[3] = t;
            t = v[6]; v[6] = v[7]; v[7] = t;
        }
        // (q7->q0): odd regs exchange across lane bit4
        v[1] = swz_xor<16>(v[1]); v[3] = swz_xor<16>(v[3]);
        v[5] = swz_xor<16>(v[5]); v[7] = swz_xor<16>(v[7]);
        // RY layer (shared theta)
        ry_lane<16>(v, lane, ct[0], st[0]);
        ry_lane<8 >(v, lane, ct[1], st[1]);
        ry_lane<4 >(v, lane, ct[2], st[2]);
        ry_lane<2 >(v, lane, ct[3], st[3]);
        ry_lane<1 >(v, lane, ct[4], st[4]);
        ry_reg<4>(v, ct[5], st[5]);
        ry_reg<2>(v, ct[6], st[6]);
        ry_reg<1>(v, ct[7], st[7]);
    }
}

// =====================================================================================
// Setup: build M = U^T (U = L^2, the fixed circuit operator) as f16 WMMA B-fragments.
// Fragment layout (per v_wmma_f32_16x16x32_f16, wave32): fragment (t,c):
//   lane l holds col n = 16t + (l&15); element e <-> K = 32c + 8*(l>>4) + (e&7) + 16*(e>>3).
// B[k][n] = U[n][k]. Wave j computes column j of U (row k=j of B) and scatters it.
// M half-index: ((t*8 + c)*32 + h*16 + nn)*16 + e
// =====================================================================================
__global__ __launch_bounds__(256) void build_op(const float* __restrict__ theta,
                                                _Float16* __restrict__ M) {
    const int lane = threadIdx.x & 31;
    const int wave = threadIdx.x >> 5;
    const int j = blockIdx.x * 8 + wave;          // basis column 0..255

    float ct[8], st[8];
    #pragma unroll
    for (int i = 0; i < 8; ++i) {
        float th = 0.5f * theta[i];
        ct[i] = __cosf(th);
        st[i] = __sinf(th);
    }

    float v[8];
    #pragma unroll
    for (int r = 0; r < 8; ++r) v[r] = (((lane << 3) | r) == j) ? 1.0f : 0.0f;

    apply_layers(v, lane, ct, st);

    const int cc = j >> 5;
    const int hh = (j >> 3) & 1;
    const int ee = (j & 7) | (((j >> 4) & 1) << 3);
    #pragma unroll
    for (int r = 0; r < 8; ++r) {
        const int n  = (lane << 3) | r;           // output state index
        const int t  = n >> 4;
        const int nn = n & 15;
        M[((size_t)(t * 8 + cc) * 32 + hh * 16 + nn) * 16 + ee] = (_Float16)v[r];
    }
}

// =====================================================================================
// Main fused kernel: LN+projection (WMMA) -> angles -> init product state (A-fragments)
// -> circuit as 16x(8xWMMA) GEMM against M -> folded readout.
// =====================================================================================
__global__ __launch_bounds__(256) void qsco_fused(
    const float* __restrict__ E, const float* __restrict__ gma, const float* __restrict__ bta,
    const float* __restrict__ Wp, const float* __restrict__ bpre,
    const float* __restrict__ Wpost, const float* __restrict__ bpost,
    const _Float16* __restrict__ M,
    float* __restrict__ out, int ntiles)
{
    __shared__ __align__(32) _Float16 bfrag[16][32][16];   // projection B fragments (16 KB)
    __shared__ float sTq[8], sCq[8];
    __shared__ float csL[8][16][8];                        // cos(ang/2) per wave/row/qubit
    __shared__ float snL[8][16][8];                        // sin(ang/2)

    const int lane = threadIdx.x & 31;
    const int wave = threadIdx.x >> 5;

    // ---------------- block setup ----------------
    // cols 0..7 = gamma .* W_pre rows, col 8 = ones (row-sum), cols 9..15 = 0
    for (int combo = threadIdx.x; combo < 512; combo += 256) {
        const int s = combo >> 5, l = combo & 31;
        const int n = l & 15, h = l >> 4;
        const int kb = 32 * s + 8 * h;
        #pragma unroll
        for (int e = 0; e < 16; ++e) {
            const int k = kb + (e & 7) + ((e & 8) ? 16 : 0);
            float val = (n < 8) ? (gma[k] * Wp[n * IN_DIM + k]) : ((n == 8) ? 1.0f : 0.0f);
            bfrag[s][l][e] = (_Float16)val;
        }
    }
    {   // T_q = sum_j gamma_j*W[q][j];  C_q = sum_j beta_j*W[q][j] + b_pre[q]
        const int q = wave;
        float t = 0.f, c = 0.f;
        for (int j = lane; j < IN_DIM; j += 32) {
            float w = Wp[q * IN_DIM + j];
            t = fmaf(gma[j], w, t);
            c = fmaf(bta[j], w, c);
        }
        t = wave_sum32(t);
        c = wave_sum32(c);
        if (lane == 0) { sTq[q] = t; sCq[q] = c + bpre[q]; }
    }
    __syncthreads();

    // ---------------- per-wave constants ----------------
    const int   ql = lane & 7;
    const float Tl = sTq[ql];
    const float Cl = sCq[ql];
    const float bp = bpost[0];

    // readout coefficients folded with W_post: coefT[t] = sum_i sign_i(n)*W_post[i],
    // n = 16t + (lane&15), qubit i <-> bit 7-i of n
    float wpo[8];
    #pragma unroll
    for (int i = 0; i < 8; ++i) wpo[i] = Wpost[i];
    float coefT[16];
    #pragma unroll
    for (int t = 0; t < 16; ++t) {
        const int n = 16 * t + (lane & 15);
        float a = 0.f;
        #pragma unroll
        for (int i = 0; i < 8; ++i) a += ((n >> (7 - i)) & 1) ? -wpo[i] : wpo[i];
        coefT[t] = a;
    }

    const int tile = blockIdx.x * 8 + wave;
    if (tile < ntiles) {
        const int R = tile * 16;
        const int h = lane >> 4;
        const float* erow = E + (size_t)(R + (lane & 15)) * IN_DIM + 8 * h;

        // ---------------- stage 1: WMMA projection (S_q, row-sum) + VALU sumsq ----------------
        v8f acc = {0.f, 0.f, 0.f, 0.f, 0.f, 0.f, 0.f, 0.f};
        float sumsq = 0.f;
        #pragma unroll
        for (int s = 0; s < 16; ++s) {
            const float* p = erow + 32 * s;
            float4 c0 = *(const float4*)(p);
            float4 c1 = *(const float4*)(p + 4);
            float4 c2 = *(const float4*)(p + 16);
            float4 c3 = *(const float4*)(p + 20);
            sumsq = fmaf(c0.x, c0.x, sumsq); sumsq = fmaf(c0.y, c0.y, sumsq);
            sumsq = fmaf(c0.z, c0.z, sumsq); sumsq = fmaf(c0.w, c0.w, sumsq);
            sumsq = fmaf(c1.x, c1.x, sumsq); sumsq = fmaf(c1.y, c1.y, sumsq);
            sumsq = fmaf(c1.z, c1.z, sumsq); sumsq = fmaf(c1.w, c1.w, sumsq);
            sumsq = fmaf(c2.x, c2.x, sumsq); sumsq = fmaf(c2.y, c2.y, sumsq);
            sumsq = fmaf(c2.z, c2.z, sumsq); sumsq = fmaf(c2.w, c2.w, sumsq);
            sumsq = fmaf(c3.x, c3.x, sumsq); sumsq = fmaf(c3.y, c3.y, sumsq);
            sumsq = fmaf(c3.z, c3.z, sumsq); sumsq = fmaf(c3.w, c3.w, sumsq);
            v16h a;
            a[0]  = (_Float16)c0.x; a[1]  = (_Float16)c0.y; a[2]  = (_Float16)c0.z; a[3]  = (_Float16)c0.w;
            a[4]  = (_Float16)c1.x; a[5]  = (_Float16)c1.y; a[6]  = (_Float16)c1.z; a[7]  = (_Float16)c1.w;
            a[8]  = (_Float16)c2.x; a[9]  = (_Float16)c2.y; a[10] = (_Float16)c2.z; a[11] = (_Float16)c2.w;
            a[12] = (_Float16)c3.x; a[13] = (_Float16)c3.y; a[14] = (_Float16)c3.z; a[15] = (_Float16)c3.w;
            v16h b = *(const v16h*)(&bfrag[s][lane][0]);
            acc = __builtin_amdgcn_wmma_f32_16x16x32_f16(false, a, false, b, (short)0, acc, false, false);
        }
        const float ssp = sumsq + swz_xor<16>(sumsq);   // full row sumsq in lanes l and l+16

        // ---------------- stage 2: per-row angles -> sincos staged to LDS ----------------
        #pragma unroll
        for (int m = 0; m < 16; ++m) {
            const int sb = (m & 8) ? 16 : 0;
            float Sq   = __shfl(acc[m & 7], (lane & 7) + sb, 32);
            float ssum = __shfl(acc[m & 7], 8 + sb, 32);
            float ss   = __shfl(ssp, m, 32);
            float mu   = ssum * (1.0f / 512.0f);
            float var  = fmaf(ss, 1.0f / 512.0f, -mu * mu);
            float rstd = rsqrtf(var + LN_EPS);
            float ang  = fmaf(rstd, fmaf(-mu, Tl, Sq), Cl);
            float ha   = 0.5f * ang;
            float cc   = __cosf(ha);
            float sn   = __sinf(ha);
            if (lane < 8) { csL[wave][m][lane] = cc; snL[wave][m][lane] = sn; }
        }

        // ---------------- stage 3: init product state as A-fragments ----------------
        // amp_m(k) = f_hi[k>>4] * f_lo[k&15] ; qubit i <-> bit 7-i of k
        const int mrow = lane & 15;
        float fc[8], fs[8];
        #pragma unroll
        for (int q = 0; q < 8; ++q) { fc[q] = csL[wave][mrow][q]; fs[q] = snL[wave][mrow][q]; }

        float t01[4], t23[4], t45[4], t67[4];
        #pragma unroll
        for (int b2 = 0; b2 < 4; ++b2) {
            t01[b2] = ((b2 & 2) ? fs[0] : fc[0]) * ((b2 & 1) ? fs[1] : fc[1]);
            t23[b2] = ((b2 & 2) ? fs[2] : fc[2]) * ((b2 & 1) ? fs[3] : fc[3]);
            t45[b2] = ((b2 & 2) ? fs[4] : fc[4]) * ((b2 & 1) ? fs[5] : fc[5]);
            t67[b2] = ((b2 & 2) ? fs[6] : fc[6]) * ((b2 & 1) ? fs[7] : fc[7]);
        }
        float f_hi[16], f_lo[16];
        #pragma unroll
        for (int g = 0; g < 16; ++g) {
            f_hi[g] = t01[g >> 2] * t23[g & 3];
            f_lo[g] = t45[g >> 2] * t67[g & 3];
        }
        // half-dependent low-factor slice (keeps register indexing static)
        float flo[8];
        #pragma unroll
        for (int x = 0; x < 8; ++x) flo[x] = h ? f_lo[8 + x] : f_lo[x];

        v16h afr[8];
        #pragma unroll
        for (int c = 0; c < 8; ++c) {
            v16h a;
            #pragma unroll
            for (int e = 0; e < 8; ++e)  a[e]     = (_Float16)(f_hi[2 * c]     * flo[e]);
            #pragma unroll
            for (int e = 0; e < 8; ++e)  a[8 + e] = (_Float16)(f_hi[2 * c + 1] * flo[e]);
            afr[c] = a;
        }

        // ---------------- stage 4: circuit as GEMM against M + folded readout ----------------
        float rowacc[8] = {0.f, 0.f, 0.f, 0.f, 0.f, 0.f, 0.f, 0.f};
        #pragma unroll
        for (int t = 0; t < 16; ++t) {
            v8f d = {0.f, 0.f, 0.f, 0.f, 0.f, 0.f, 0.f, 0.f};
            #pragma unroll
            for (int c = 0; c < 8; ++c) {
                v16h b = *(const v16h*)(M + ((size_t)(t * 8 + c) * 32 + lane) * 16);
                d = __builtin_amdgcn_wmma_f32_16x16x32_f16(false, afr[c], false, b, (short)0, d, false, false);
            }
            const float w = coefT[t];
            #pragma unroll
            for (int r = 0; r < 8; ++r) rowacc[r] = fmaf(d[r] * d[r], w, rowacc[r]);
        }
        // reduce over the 16 lanes of each half (lanes 0-15 -> rows 0-7, lanes 16-31 -> rows 8-15)
        #pragma unroll
        for (int r = 0; r < 8; ++r) {
            float x = rowacc[r];
            x += swz_xor<1>(x);
            x += swz_xor<2>(x);
            x += swz_xor<4>(x);
            x += swz_xor<8>(x);
            rowacc[r] = x + bp;
        }
        if (lane == 0) {
            float4 o0 = {rowacc[0], rowacc[1], rowacc[2], rowacc[3]};
            float4 o1 = {rowacc[4], rowacc[5], rowacc[6], rowacc[7]};
            *(float4*)(out + R)     = o0;
            *(float4*)(out + R + 4) = o1;
        } else if (lane == 16) {
            float4 o0 = {rowacc[0], rowacc[1], rowacc[2], rowacc[3]};
            float4 o1 = {rowacc[4], rowacc[5], rowacc[6], rowacc[7]};
            *(float4*)(out + R + 8)  = o0;
            *(float4*)(out + R + 12) = o1;
        }
    }
}

extern "C" void kernel_launch(void* const* d_in, const int* in_sizes, int n_in,
                              void* d_out, int out_size, void* d_ws, size_t ws_size,
                              hipStream_t stream) {
    const float* E     = (const float*)d_in[0];
    const float* gma   = (const float*)d_in[1];
    const float* bta   = (const float*)d_in[2];
    const float* Wp    = (const float*)d_in[3];
    const float* bpre  = (const float*)d_in[4];
    const float* theta = (const float*)d_in[5];
    const float* Wpost = (const float*)d_in[6];
    const float* bpost = (const float*)d_in[7];
    float* out = (float*)d_out;
    _Float16* M = (_Float16*)d_ws;               // 256x256 f16 fragments = 128 KB

    build_op<<<32, 256, 0, stream>>>(theta, M);

    const int B = in_sizes[0] / IN_DIM;
    const int ntiles = B / 16;                   // B=131072 -> 8192 tiles
    const int blocks = (ntiles + 7) / 8;         // 8 waves per block, 1 tile per wave
    qsco_fused<<<blocks, 256, 0, stream>>>(E, gma, bta, Wp, bpre, Wpost, bpost, M, out, ntiles);
}
